// XMCBCECHUNKEDLayer_37340445672285
// MI455X (gfx1250) — compile-verified, matrix operand order checked by reading, not run on hardware
//
#include <hip/hip_runtime.h>
#include <hip/hip_bf16.h>

// ---------------------------------------------------------------------------
// Fused XMC BCE forward/backward for MI455X (gfx1250, wave32, WMMA).
//   logits = E @ W^T ; p = sigmoid(logits) ; p[rows,cols] -= 1
//   grad_input = p @ W          (register-accumulated, atomically reduced)
//   W_new = W*(1-lr*wd) - lr*(p^T @ E)
// Single streaming pass over W (HBM floor: ~806 MB => ~35 us at 23.3 TB/s).
// All GEMMs in bf16 WMMA 16x16x32 with f32 accumulation. Dual LDS layouts
// (row-major + transposed) keep every WMMA fragment load K-contiguous
// (2x ds_load_b128 per fragment); fragments are preloaded per chunk so the
// WMMA burst issues behind a single s_wait_dscnt.
// ---------------------------------------------------------------------------

typedef __bf16 bf16_t;
typedef __attribute__((ext_vector_type(16))) __bf16 v16bf;
typedef __attribute__((ext_vector_type(8)))  __bf16 v8bf;
typedef __attribute__((ext_vector_type(8)))  float  v8f;

#define B_DIM   128
#define D_DIM   768
#define L_DIM   131072
#define WD_CONST 1.0e-4f

#define LTILE   128                    // L rows per tile
#define NTILES_L (L_DIM / LTILE)       // 1024
#define KCHUNK  32                     // K step for WMMA (bf16 16x16x32)
#define NCHUNKS (D_DIM / KCHUNK)       // 24
#define THREADS 512                    // 16 waves of 32
#define GRID    256                    // each WG handles 4 L tiles

// ---- WMMA wrapper ----------------------------------------------------------
__device__ __forceinline__ v8f wmma_bf16(v16bf a, v16bf b, v8f c) {
  return __builtin_amdgcn_wmma_f32_16x16x32_bf16(
      false, a, false, b, (short)0, c, false, false);
}

// ---- K-contiguous fragment builders (2x ds_load_b128 each) -----------------
// A fragment: element (m,k) at lds[(row0+m)*pitch + k0 + k]
// lanes 0-15: M=0..15, K in [kb,kb+8) and [kb+16,kb+24), kb = 8*(lane>>4)
__device__ __forceinline__ v16bf load_frag_a_c(const bf16_t* lds, int row0,
                                               int pitch, int k0) {
  const int l  = threadIdx.x & 31;
  const bf16_t* base = lds + (row0 + (l & 15)) * pitch + k0 + ((l >> 4) << 3);
  v8bf lo = *(const v8bf*)(base);
  v8bf hi = *(const v8bf*)(base + 16);
  return __builtin_shufflevector(lo, hi, 0,1,2,3,4,5,6,7,8,9,10,11,12,13,14,15);
}

// B fragment: element (k,n) at lds[(col0+n)*pitch + k0 + k]  (K-major storage)
__device__ __forceinline__ v16bf load_frag_b_c(const bf16_t* lds, int col0,
                                               int pitch, int k0) {
  const int l  = threadIdx.x & 31;
  const bf16_t* base = lds + (col0 + (l & 15)) * pitch + k0 + ((l >> 4) << 3);
  v8bf lo = *(const v8bf*)(base);
  v8bf hi = *(const v8bf*)(base + 16);
  return __builtin_shufflevector(lo, hi, 0,1,2,3,4,5,6,7,8,9,10,11,12,13,14,15);
}

// ---- Staging: one 128x32 f32 chunk -> LDS bf16 -----------------------------
// 512 threads: thread t handles row (t>>2), 8 cols at (t&3)*8.
__device__ __forceinline__ void stage_rm(bf16_t* __restrict__ dst,
                                         const float* __restrict__ src, int k0,
                                         const float* __restrict__ pf) {
  const int t = threadIdx.x, row = t >> 2, c8 = (t & 3) << 3;
  const float* s = src + (size_t)row * D_DIM + k0 + c8;
  float4 f0 = *(const float4*)(s);
  float4 f1 = *(const float4*)(s + 4);
  if (pf)  // stream next L tile toward L2 while we compute this one
    __builtin_prefetch(pf + (size_t)row * D_DIM + k0 + c8, 0, 1);
  v8bf v;
  v[0] = (bf16_t)f0.x; v[1] = (bf16_t)f0.y; v[2] = (bf16_t)f0.z; v[3] = (bf16_t)f0.w;
  v[4] = (bf16_t)f1.x; v[5] = (bf16_t)f1.y; v[6] = (bf16_t)f1.z; v[7] = (bf16_t)f1.w;
  *(v8bf*)(dst + row * KCHUNK + c8) = v;          // dst[row][c], pitch 32
}

__device__ __forceinline__ void stage_tr(bf16_t* __restrict__ dst,
                                         const float* __restrict__ src, int k0) {
  const int t = threadIdx.x, row = t >> 2, c8 = (t & 3) << 3;
  const float* s = src + (size_t)row * D_DIM + k0 + c8;
  float4 f0 = *(const float4*)(s);
  float4 f1 = *(const float4*)(s + 4);
  bf16_t* d = dst + c8 * LTILE + row;             // dst[c][row], pitch 128
  d[0 * LTILE] = (bf16_t)f0.x; d[1 * LTILE] = (bf16_t)f0.y;
  d[2 * LTILE] = (bf16_t)f0.z; d[3 * LTILE] = (bf16_t)f0.w;
  d[4 * LTILE] = (bf16_t)f1.x; d[5 * LTILE] = (bf16_t)f1.y;
  d[6 * LTILE] = (bf16_t)f1.z; d[7 * LTILE] = (bf16_t)f1.w;
}

// ---------------------------------------------------------------------------
__global__ void __launch_bounds__(THREADS)
xmc_fused_kernel(const float* __restrict__ embed,
                 const float* __restrict__ weight,
                 const int*   __restrict__ labels,
                 const float* __restrict__ lr_ptr,
                 float* __restrict__ grad_out,
                 float* __restrict__ weight_out)
{
  extern __shared__ char smem_raw[];
  bf16_t* ET  = (bf16_t*)smem_raw;          // embed^T  [768][128]  (192 KB)
  bf16_t* Pb  = ET + D_DIM * B_DIM;         // p        [128][128]  (32 KB)
  bf16_t* PbT = Pb + B_DIM * LTILE;         // p^T      [128][128]  (32 KB)
  bf16_t* SC  = PbT + LTILE * B_DIM;        // staging scratch 4x4096 (32 KB)
  // phase A: Ec{0,1} = SC+0/+4096 ; Wc{0,1} = SC+8192/+12288
  // phase C: WcT{0,1} = SC+0/+4096   (disjoint in time)

  const int t    = threadIdx.x;
  const int w    = t >> 5;                  // wave 0..15
  const int lane = t & 31;
  const float lr = lr_ptr[0];

  // ---- one-time: embed -> LDS transposed bf16 ------------------------------
  for (int i = t; i < (B_DIM * D_DIM) / 4; i += THREADS) {
    float4 v = ((const float4*)embed)[i];
    const int e0 = i * 4, row = e0 / D_DIM, col = e0 % D_DIM;
    ET[(col + 0) * B_DIM + row] = (bf16_t)v.x;
    ET[(col + 1) * B_DIM + row] = (bf16_t)v.y;
    ET[(col + 2) * B_DIM + row] = (bf16_t)v.z;
    ET[(col + 3) * B_DIM + row] = (bf16_t)v.w;
  }

  // persistent grad_input accumulators: wave w owns rows tile (w&7), col half
  const int mt = w & 7;
  const int nh = (w >> 3) << 4;
  v8f gacc[NCHUNKS];
#pragma unroll
  for (int i = 0; i < NCHUNKS; ++i) gacc[i] = (v8f){0,0,0,0,0,0,0,0};

  __syncthreads();

  for (int tile = blockIdx.x; tile < NTILES_L; tile += gridDim.x) {
    const int lbase = tile * LTILE;
    const float* Wg = weight + (size_t)lbase * D_DIM;
    const int tnext = tile + gridDim.x;
    const float* Wg_next =
        (tnext < NTILES_L) ? weight + (size_t)tnext * LTILE * D_DIM : nullptr;

    // ---- Phase A: logits(128x128) = E @ W_tile^T (double-buffered) --------
    v8f acc[4];
#pragma unroll
    for (int i = 0; i < 4; ++i) acc[i] = (v8f){0,0,0,0,0,0,0,0};

    stage_rm(SC + 0,    embed, 0, nullptr);       // Ec0
    stage_rm(SC + 8192, Wg,    0, Wg_next);       // Wc0
    for (int kc = 0; kc < NCHUNKS; ++kc) {
      __syncthreads();
      const int cur = kc & 1;
      if (kc + 1 < NCHUNKS) {
        stage_rm(SC + (cur ^ 1) * 4096,        embed, (kc + 1) * KCHUNK, nullptr);
        stage_rm(SC + 8192 + (cur ^ 1) * 4096, Wg,    (kc + 1) * KCHUNK, Wg_next);
      }
      // preload all fragments, then burst the WMMAs behind one dscnt wait
      v16bf bf = load_frag_b_c(SC + 8192 + cur * 4096, (w & 7) * 16, KCHUNK, 0);
      v16bf af[4];
#pragma unroll
      for (int i = 0; i < 4; ++i)
        af[i] = load_frag_a_c(SC + cur * 4096, ((w >> 3) * 4 + i) * 16, KCHUNK, 0);
#pragma unroll
      for (int i = 0; i < 4; ++i)
        acc[i] = wmma_bf16(af[i], bf, acc[i]);
    }

    // ---- Phase B: p = sigmoid(logits) -> Pb and PbT; sparse label fix -----
    __syncthreads();
#pragma unroll
    for (int i = 0; i < 4; ++i) {
      const int m0 = ((w >> 3) * 4 + i) * 16 + ((lane >> 4) << 3);
      const int n  = (w & 7) * 16 + (lane & 15);
      v8bf pv;
#pragma unroll
      for (int r = 0; r < 8; ++r) {
        const float x = acc[i][r];
        const float p = __builtin_amdgcn_rcpf(1.0f + __expf(-x));  // v_rcp_f32
        Pb[(m0 + r) * LTILE + n] = (bf16_t)p;
        pv[r] = (bf16_t)p;
      }
      *(v8bf*)(PbT + n * B_DIM + m0) = pv;        // vectorized transposed store
    }
    __syncthreads();
    {
      // 512 threads <-> 512 labels
      const int rrow = labels[2 * t];
      const int ccol = labels[2 * t + 1];
      const int cl   = ccol - lbase;
      if (cl >= 0 && cl < LTILE) {
        Pb[rrow * LTILE + cl]  = (bf16_t)((float)Pb[rrow * LTILE + cl] - 1.0f);
        PbT[cl * B_DIM + rrow] = (bf16_t)((float)PbT[cl * B_DIM + rrow] - 1.0f);
      }
    }
    __syncthreads();

    // ---- Phase C: grad_input partial += p @ W_tile  (K = 128 L rows) ------
    stage_tr(SC + 0, Wg, 0);                      // WcT0 (re-read: L2 hit)
#pragma unroll
    for (int nc = 0; nc < NCHUNKS; ++nc) {
      __syncthreads();
      const int cur = nc & 1;
      if (nc + 1 < NCHUNKS) stage_tr(SC + (cur ^ 1) * 4096, Wg, (nc + 1) * KCHUNK);
      v16bf af[4], bf[4];
#pragma unroll
      for (int k = 0; k < 4; ++k) {
        // A[m][k'] = p[m][k']             : Pb row-major, K-contiguous
        af[k] = load_frag_a_c(Pb, mt * 16, LTILE, k * KCHUNK);
        // B[k'][n] = W[k'][dcol] = WcT[dcol][k'] : pitch 128, K-contiguous
        bf[k] = load_frag_b_c(SC + cur * 4096, nh, LTILE, k * KCHUNK);
      }
#pragma unroll
      for (int k = 0; k < 4; ++k)
        gacc[nc] = wmma_bf16(af[k], bf[k], gacc[nc]);
    }

    // ---- Phase D: W_new = W*(1-lr*wd) - lr * (p^T @ E) --------------------
    const float decay = 1.0f - lr * WD_CONST;
    for (int nc = 0; nc < NCHUNKS; ++nc) {
      v16bf af[4], bf[4];
#pragma unroll
      for (int k = 0; k < 4; ++k) {
        // A[m][k'] = p[k'][lrow m] = PbT[m][k'] : pitch 128, K-contiguous
        af[k] = load_frag_a_c(PbT, mt * 16, B_DIM, k * KCHUNK);
        // B[k'][n] = E[k'][dcol] = ET[dcol][k'] : pitch 128, K-contiguous
        bf[k] = load_frag_b_c(ET, nc * KCHUNK + nh, B_DIM, k * KCHUNK);
      }
      v8f c = (v8f){0,0,0,0,0,0,0,0};
#pragma unroll
      for (int k = 0; k < 4; ++k)
        c = wmma_bf16(af[k], bf[k], c);

      const int lrow0 = lbase + mt * 16 + ((lane >> 4) << 3);
      const int dcol  = nc * KCHUNK + nh + (lane & 15);
#pragma unroll
      for (int r = 0; r < 8; ++r) {
        const size_t off = (size_t)(lrow0 + r) * D_DIM + dcol;
        weight_out[off] = weight[off] * decay - lr * c[r];
      }
    }
    __syncthreads();   // protect Pb/PbT/SC before next iteration rewrites them
  }

  // ---- Flush grad_input partials (one atomic pass per workgroup) ----------
#pragma unroll
  for (int nc = 0; nc < NCHUNKS; ++nc) {
    const int m0 = mt * 16 + ((lane >> 4) << 3);
    const int n  = nc * KCHUNK + nh + (lane & 15);
#pragma unroll
    for (int r = 0; r < 8; ++r) {
      atomicAdd(&grad_out[(m0 + r) * D_DIM + n], gacc[nc][r]);
    }
  }
}

// ---------------------------------------------------------------------------
__global__ void zero_grad_kernel(float* __restrict__ g, int n) {
  int i = blockIdx.x * blockDim.x + threadIdx.x;
  if (i < n) g[i] = 0.0f;
}

extern "C" void kernel_launch(void* const* d_in, const int* in_sizes, int n_in,
                              void* d_out, int out_size, void* d_ws, size_t ws_size,
                              hipStream_t stream) {
  const float* embed  = (const float*)d_in[0];
  const float* weight = (const float*)d_in[1];
  const int*   labels = (const int*)d_in[2];
  const float* lr     = (const float*)d_in[3];

  float* grad_out   = (float*)d_out;                 // 128*768
  float* weight_out = grad_out + B_DIM * D_DIM;      // 131072*768

  zero_grad_kernel<<<(B_DIM * D_DIM + 255) / 256, 256, 0, stream>>>(
      grad_out, B_DIM * D_DIM);

  // ET (768*128) + Pb (128*128) + PbT (128*128) + scratch (4*4096), bf16
  const size_t shmem =
      (size_t)(D_DIM * B_DIM + 2 * B_DIM * LTILE + 4 * 4096) * sizeof(bf16_t);
  xmc_fused_kernel<<<dim3(GRID), dim3(THREADS), shmem, stream>>>(
      embed, weight, labels, lr, grad_out, weight_out);
}
